// Attention_34918084116775
// MI455X (gfx1250) — matrix operand order repaired and checked by reading, not compiled, not run on hardware
//
#include <hip/hip_runtime.h>

// Problem constants
#define BB   2
#define SS   2048
#define HH   2048
#define NHH  16
#define HDD  128
#define MM   (BB * SS)          // 4096 rows in all projection GEMMs

typedef __attribute__((ext_vector_type(16))) __bf16 v16bf;
typedef __attribute__((ext_vector_type(8)))  float  v8f;

union BF16Frag { v16bf v; unsigned int u[8]; };

__device__ __forceinline__ unsigned short f2bf(float f) {
    unsigned int u = __float_as_uint(f);
    u += 0x7FFFu + ((u >> 16) & 1u);          // round-to-nearest-even
    return (unsigned short)(u >> 16);
}

// gfx1250 async DMA: global -> LDS, 16B per lane, tracked by ASYNCcnt.
// lds_off = byte offset of destination within the workgroup LDS allocation.
__device__ __forceinline__ void async_copy_b128(unsigned lds_off, const void* gptr) {
    asm volatile("global_load_async_to_lds_b128 %0, %1, off"
                 :: "v"(lds_off), "v"(gptr) : "memory");
}
__device__ __forceinline__ void wait_async_all() {
    asm volatile("s_wait_asynccnt 0" ::: "memory");
}

// ---------------------------------------------------------------------------
// Kernel 1: f32 -> bf16 cast (x and weights)
// ---------------------------------------------------------------------------
__global__ __launch_bounds__(256)
void cast_f32_bf16(const float* __restrict__ in, unsigned short* __restrict__ out, int n) {
    int i = blockIdx.x * blockDim.x + threadIdx.x;
    if (i < n) out[i] = f2bf(in[i]);
}

// ---------------------------------------------------------------------------
// Kernel 2: bf16 WMMA GEMM: out[m,n] = sum_k A[m,k] * W[n,k] + bias[n]
//   A: [M][K] bf16 row-major, W: [N][K] bf16 row-major (i.e. B = W^T)
//   Block tile 128x64, K-step 32, 4 waves; wave w -> rows w*32 .. w*32+31
//   (2 A-frags x 4 B-frags = 8 WMMA per K-step, B-frags reused across M).
//   Tiles staged with GLOBAL_LOAD_ASYNC_TO_LDS_B128 (ASYNCcnt).
//   EPI 0: f32 out, plain [M][N] (final projection -> d_out)
//   EPI 1: bias + RoPE, bf16 out in [B,NH,S,HD]   (Q, K)
//   EPI 2: bias,        bf16 out in [B,NH,S,HD]   (V)
// ---------------------------------------------------------------------------
template <int EPI>
__global__ __launch_bounds__(128)
void gemm_bf16(const unsigned short* __restrict__ Abf,
               const unsigned short* __restrict__ Wbf,
               const float* __restrict__ bias,
               float* __restrict__ outF,
               unsigned short* __restrict__ outB,
               const float* __restrict__ fcos,
               const float* __restrict__ fsin) {
    // Row stride 20 u32 (80 B): keeps 16B alignment for b128 async stores and
    // spreads rows over distinct LDS banks (20*r mod 64 distinct for r<16).
    __shared__ __align__(16) unsigned int lA[128 * 20];
    __shared__ __align__(16) unsigned int lB[64 * 20];

    const unsigned int* a32 = (const unsigned int*)Abf;
    const unsigned int* w32 = (const unsigned int*)Wbf;

    const int m0   = blockIdx.y * 128;
    const int n0   = blockIdx.x * 64;
    const int tid  = threadIdx.x;
    const int wm   = tid >> 5;          // wave id 0..3
    const int lane = tid & 31;
    const int r    = lane & 15;
    const int half = lane >> 4;

    v8f acc[2][4];
#pragma unroll
    for (int i = 0; i < 2; ++i)
#pragma unroll
        for (int j = 0; j < 4; ++j) acc[i][j] = (v8f){0, 0, 0, 0, 0, 0, 0, 0};

    for (int k0 = 0; k0 < HH; k0 += 32) {
        // ---- async-stage A (128x32 bf16) and B (64x32 bf16) tiles ----
#pragma unroll
        for (int j = 0; j < 4; ++j) {            // A: 512 b128 chunks
            int idx = j * 128 + tid;             // 0..511
            int row = idx >> 2, g = idx & 3;     // 4 x 16B groups per row
            async_copy_b128((unsigned)(uintptr_t)&lA[row * 20 + g * 4],
                            &a32[(m0 + row) * (HH / 2) + (k0 >> 1) + g * 4]);
        }
#pragma unroll
        for (int j = 0; j < 2; ++j) {            // B: 256 b128 chunks
            int idx = j * 128 + tid;             // 0..255
            int row = idx >> 2, g = idx & 3;
            async_copy_b128((unsigned)(uintptr_t)&lB[row * 20 + g * 4],
                            &w32[(n0 + row) * (HH / 2) + (k0 >> 1) + g * 4]);
        }
        wait_async_all();
        __syncthreads();

        // ---- gather 2 A fragments (16x32 bf16 each, CDNA5 A layout) ----
        BF16Frag af[2];
#pragma unroll
        for (int i = 0; i < 2; ++i)
#pragma unroll
            for (int v = 0; v < 8; ++v) {
                int p = (v & 3) + 4 * half + 8 * (v >> 2);   // K-pair index
                af[i].u[v] = lA[(wm * 32 + i * 16 + r) * 20 + p];
            }

        // ---- 4 N-tiles: gather B fragment (32x16), 2 WMMAs each ----
#pragma unroll
        for (int nt = 0; nt < 4; ++nt) {
            BF16Frag bfg;
#pragma unroll
            for (int v = 0; v < 8; ++v)
                bfg.u[v] = lB[(nt * 16 + r) * 20 + ((v & 3) + 4 * half + 8 * (v >> 2))];
            acc[0][nt] = __builtin_amdgcn_wmma_f32_16x16x32_bf16(
                false, af[0].v, false, bfg.v, (short)0, acc[0][nt], false, false);
            acc[1][nt] = __builtin_amdgcn_wmma_f32_16x16x32_bf16(
                false, af[1].v, false, bfg.v, (short)0, acc[1][nt], false, false);
        }
        __syncthreads();
    }

    // ---- epilogue ----
#pragma unroll
    for (int i = 0; i < 2; ++i) {
#pragma unroll
        for (int nt = 0; nt < 4; ++nt) {
#pragma unroll
            for (int rr = 0; rr < 8; ++rr) {
                int row = m0 + wm * 32 + i * 16 + rr + 8 * half;   // m index
                int col = n0 + nt * 16 + r;                        // n index
                float val = acc[i][nt][rr] + bias[col];
                if (EPI == 0) {
                    outF[row * HH + col] = val;
                } else {
                    int b = row >> 11;            // row / S
                    int s = row & (SS - 1);
                    int h = col >> 7;             // col / HD
                    int d = col & (HDD - 1);
                    float res = val;
                    if (EPI == 1) {
                        // RoPE: partner feature (d^1) lives in lane^1 (same row)
                        float c  = fcos[s * HDD + d];
                        float sn = fsin[s * HDD + d];
                        float pv = __shfl_xor(val, 1, 32);
                        res = (d & 1) ? (val * c + pv * sn) : (val * c - pv * sn);
                    }
                    outB[((b * NHH + h) * SS + s) * HDD + d] = f2bf(res);
                }
            }
        }
    }
}

// ---------------------------------------------------------------------------
// Kernel 3: flash attention, one wave per (b, h, 16-query tile).
//   Streams keys in chunks of 32 (two 16x16 score tiles), online softmax,
//   O accumulated as 8 f32 C-tiles (16 x 128). P goes through LDS to be
//   re-laid-out from C-format f32 into A-format bf16.
// ---------------------------------------------------------------------------
__global__ __launch_bounds__(32)
void flash_attn(const unsigned short* __restrict__ qb,
                const unsigned short* __restrict__ kb,
                const unsigned short* __restrict__ vb,
                unsigned short* __restrict__ ob) {
    __shared__ unsigned int ldsP[16 * 16];   // 16 queries x 32 keys (bf16)

    const int bh   = blockIdx.y;             // b*NH + h
    const int q0   = blockIdx.x * 16;
    const int lane = threadIdx.x & 31;
    const int r    = lane & 15;
    const int half = lane >> 4;

    const unsigned int*   q32 = (const unsigned int*)(qb + (size_t)bh * SS * HDD);
    const unsigned int*   k32 = (const unsigned int*)(kb + (size_t)bh * SS * HDD);
    const unsigned short* vB  = vb + (size_t)bh * SS * HDD;

    // Q tile as 4 A-fragments (K = HD in chunks of 32)
    BF16Frag qf[4];
#pragma unroll
    for (int kc = 0; kc < 4; ++kc)
#pragma unroll
        for (int v = 0; v < 8; ++v) {
            int p = (v & 3) + 4 * half + 8 * (v >> 2);
            qf[kc].u[v] = q32[(q0 + r) * 64 + kc * 16 + p];
        }

    v8f O[8];
#pragma unroll
    for (int i = 0; i < 8; ++i) O[i] = (v8f){0, 0, 0, 0, 0, 0, 0, 0};
    float mrun[8], lrun[8];
#pragma unroll
    for (int i = 0; i < 8; ++i) { mrun[i] = -3.0e38f; lrun[i] = 0.0f; }

    const int   qmax  = q0 + 15;
    const float scale = 0.0883883476483184f;   // 1/sqrt(128)

    for (int j0 = 0; j0 <= qmax; j0 += 32) {
        float sv[2][8];

        // ---- scores: two 16x16 tiles of Q . K^T ----
#pragma unroll
        for (int jt = 0; jt < 2; ++jt) {
            int kt = j0 + jt * 16;
            v8f a = (v8f){0, 0, 0, 0, 0, 0, 0, 0};
#pragma unroll
            for (int kc = 0; kc < 4; ++kc) {
                BF16Frag bfg;   // B = K^T: column = key (lane), K dim pairs contiguous
#pragma unroll
                for (int v = 0; v < 8; ++v)
                    bfg.u[v] = k32[(kt + r) * 64 + kc * 16 + ((v & 3) + 4 * half + 8 * (v >> 2))];
                a = __builtin_amdgcn_wmma_f32_16x16x32_bf16(
                    false, qf[kc].v, false, bfg.v, (short)0, a, false, false);
            }
            int kj = kt + r;                       // key index of this lane's column
#pragma unroll
            for (int rr = 0; rr < 8; ++rr) {
                int qi = q0 + rr + 8 * half;
                sv[jt][rr] = (kj > qi) ? -3.0e38f : a[rr] * scale;
            }
        }

        // ---- online softmax over the 32 new keys ----
        float fsc[8];
#pragma unroll
        for (int rr = 0; rr < 8; ++rr) {
            float mt = fmaxf(sv[0][rr], sv[1][rr]);
            mt = fmaxf(mt, __shfl_xor(mt, 1, 32));
            mt = fmaxf(mt, __shfl_xor(mt, 2, 32));
            mt = fmaxf(mt, __shfl_xor(mt, 4, 32));
            mt = fmaxf(mt, __shfl_xor(mt, 8, 32));
            float nm = fmaxf(mrun[rr], mt);
            fsc[rr]  = __expf(mrun[rr] - nm);
            mrun[rr] = nm;
            float p0 = __expf(sv[0][rr] - nm);
            float p1 = __expf(sv[1][rr] - nm);
            sv[0][rr] = p0; sv[1][rr] = p1;
            float sum = p0 + p1;
            sum += __shfl_xor(sum, 1, 32);
            sum += __shfl_xor(sum, 2, 32);
            sum += __shfl_xor(sum, 4, 32);
            sum += __shfl_xor(sum, 8, 32);
            lrun[rr] = lrun[rr] * fsc[rr] + sum;
        }
#pragma unroll
        for (int nt = 0; nt < 8; ++nt)
#pragma unroll
            for (int rr = 0; rr < 8; ++rr) O[nt][rr] *= fsc[rr];

        // ---- C-layout f32 P -> LDS -> A-layout bf16 fragment ----
        unsigned short* lp16 = (unsigned short*)ldsP;
#pragma unroll
        for (int jt = 0; jt < 2; ++jt)
#pragma unroll
            for (int rr = 0; rr < 8; ++rr)
                lp16[(rr + 8 * half) * 32 + jt * 16 + r] = f2bf(sv[jt][rr]);
        __syncthreads();
        BF16Frag pf;
#pragma unroll
        for (int v = 0; v < 8; ++v) {
            int p = (v & 3) + 4 * half + 8 * (v >> 2);
            pf.u[v] = ldsP[r * 16 + p];
        }
        __syncthreads();

        // ---- O += P . V  (8 n-tiles of 16, K = 32 keys) ----
#pragma unroll
        for (int nt = 0; nt < 8; ++nt) {
            BF16Frag vf;   // B[key][dim]; key pairs stride HD apart -> pack 2 u16
#pragma unroll
            for (int v = 0; v < 8; ++v) {
                int p = (v & 3) + 4 * half + 8 * (v >> 2);
                unsigned int lo = vB[(j0 + 2 * p)     * HDD + nt * 16 + r];
                unsigned int hi = vB[(j0 + 2 * p + 1) * HDD + nt * 16 + r];
                vf.u[v] = lo | (hi << 16);
            }
            O[nt] = __builtin_amdgcn_wmma_f32_16x16x32_bf16(
                false, pf.v, false, vf.v, (short)0, O[nt], false, false);
        }
    }

    // ---- normalize and store attn output as bf16 in [B, S, H] layout ----
    const int b = bh >> 4, h = bh & 15;
#pragma unroll
    for (int rr = 0; rr < 8; ++rr) {
        float inv = 1.0f / lrun[rr];
        int qi = q0 + rr + 8 * half;
#pragma unroll
        for (int nt = 0; nt < 8; ++nt) {
            float val = O[nt][rr] * inv;
            ob[(size_t)(b * SS + qi) * HH + h * HDD + nt * 16 + r] = f2bf(val);
        }
    }
}

// ---------------------------------------------------------------------------
// Host-side launch
// ---------------------------------------------------------------------------
extern "C" void kernel_launch(void* const* d_in, const int* in_sizes, int n_in,
                              void* d_out, int out_size, void* d_ws, size_t ws_size,
                              hipStream_t stream) {
    const float* x    = (const float*)d_in[0];
    const float* fcos = (const float*)d_in[1];
    const float* fsin = (const float*)d_in[2];
    const float* Wq   = (const float*)d_in[3];
    const float* bq   = (const float*)d_in[4];
    const float* Wk   = (const float*)d_in[5];
    const float* bk   = (const float*)d_in[6];
    const float* Wv   = (const float*)d_in[7];
    const float* bv   = (const float*)d_in[8];
    const float* Wo   = (const float*)d_in[9];
    const float* bo   = (const float*)d_in[10];
    // d_in[11] = attn_mask: causal, handled analytically in flash_attn

    char* ws = (char*)d_ws;
    const size_t MB = 1024ull * 1024ull;
    unsigned short* xb    = (unsigned short*)(ws);              // 16 MB
    unsigned short* Wqb   = (unsigned short*)(ws + 16 * MB);    //  8 MB
    unsigned short* Wkb   = (unsigned short*)(ws + 24 * MB);    //  8 MB
    unsigned short* Wvb   = (unsigned short*)(ws + 32 * MB);    //  8 MB
    unsigned short* Wob   = (unsigned short*)(ws + 40 * MB);    //  8 MB
    unsigned short* qbf   = (unsigned short*)(ws + 48 * MB);    // 16 MB
    unsigned short* kbf   = (unsigned short*)(ws + 64 * MB);    // 16 MB
    unsigned short* vbf   = (unsigned short*)(ws + 80 * MB);    // 16 MB
    unsigned short* attnb = (unsigned short*)(ws + 96 * MB);    // 16 MB  (112 MB total)

    const int nx = MM * HH;       // 8,388,608
    const int nw = HH * HH;       // 4,194,304
    cast_f32_bf16<<<(nx + 255) / 256, 256, 0, stream>>>(x,  xb,  nx);
    cast_f32_bf16<<<(nw + 255) / 256, 256, 0, stream>>>(Wq, Wqb, nw);
    cast_f32_bf16<<<(nw + 255) / 256, 256, 0, stream>>>(Wk, Wkb, nw);
    cast_f32_bf16<<<(nw + 255) / 256, 256, 0, stream>>>(Wv, Wvb, nw);
    cast_f32_bf16<<<(nw + 255) / 256, 256, 0, stream>>>(Wo, Wob, nw);

    dim3 gg(HH / 64, MM / 128);   // 32 x 32 blocks, 128 threads
    gemm_bf16<1><<<gg, 128, 0, stream>>>(xb, Wqb, bq, nullptr, qbf, fcos, fsin);
    gemm_bf16<1><<<gg, 128, 0, stream>>>(xb, Wkb, bk, nullptr, kbf, fcos, fsin);
    gemm_bf16<2><<<gg, 128, 0, stream>>>(xb, Wvb, bv, nullptr, vbf, nullptr, nullptr);

    flash_attn<<<dim3(SS / 16, BB * NHH), 32, 0, stream>>>(qbf, kbf, vbf, attnb);

    gemm_bf16<0><<<gg, 128, 0, stream>>>(attnb, Wob, bo, (float*)d_out, nullptr,
                                         nullptr, nullptr);
}
